// WebObjExtractionNet_34153579938092
// MI455X (gfx1250) — matrix operand order repaired and checked by reading, not compiled, not run on hardware
//
#include <hip/hip_runtime.h>
#include <hip/hip_bf16.h>
#include <math.h>

typedef __attribute__((ext_vector_type(16))) __bf16 bf16x16;
typedef __attribute__((ext_vector_type(8)))  __bf16 bf16x8;
typedef __attribute__((ext_vector_type(8)))  float  f32x8;

#define OUT3   3
#define HEADS  8
#define HID    384
#define NCLS   10
#define CFEAT  384
#define NVIS   (CFEAT*OUT3*OUT3)   /* 3456 */
#define BBH    32
#define NOWN   (NVIS+BBH)          /* 3488 */
#define N_BOX  1024
#define N_CTXI 16
#define NCOMB  (HEADS*NOWN)        /* 27904 */
#define NDEC   (2*NOWN)            /* 6976  */

__device__ __forceinline__ int imin(int a, int b){ return a < b ? a : b; }
__device__ __forceinline__ int imax(int a, int b){ return a > b ? a : b; }

// ---------------- CNN backbone (f32 direct, ~1% of FLOPs) ----------------

__global__ void conv2d_bias(const float* __restrict__ in, const float* __restrict__ w,
                            const float* __restrict__ bias, float* __restrict__ out,
                            int N, int Cin, int Hin, int Win, int Cout, int Hout, int Wout,
                            int Kh, int Kw, int stride, int pad, int doRelu)
{
    int idx = blockIdx.x * blockDim.x + threadIdx.x;
    int total = N * Cout * Hout * Wout;
    if (idx >= total) return;
    int x = idx % Wout; int t = idx / Wout;
    int y = t % Hout;   t /= Hout;
    int o = t % Cout;   int n = t / Cout;
    float acc = bias[o];
    int iy0 = y * stride - pad, ix0 = x * stride - pad;
    const float* wbase = w + (size_t)o * Cin * Kh * Kw;
    const float* ibase = in + (size_t)n * Cin * Hin * Win;
    for (int c = 0; c < Cin; c++) {
        const float* ip = ibase + (size_t)c * Hin * Win;
        const float* wp = wbase + (size_t)c * Kh * Kw;
        for (int ky = 0; ky < Kh; ky++) {
            int iy = iy0 + ky;
            if ((unsigned)iy >= (unsigned)Hin) continue;
            for (int kx = 0; kx < Kw; kx++) {
                int ix = ix0 + kx;
                if ((unsigned)ix >= (unsigned)Win) continue;
                acc += ip[iy * Win + ix] * wp[ky * Kw + kx];
            }
        }
    }
    if (doRelu) acc = fmaxf(acc, 0.0f);
    out[idx] = acc;
}

__global__ void maxpool3x3s2(const float* __restrict__ in, float* __restrict__ out,
                             int NC, int Hin, int Win, int Hout, int Wout)
{
    int idx = blockIdx.x * blockDim.x + threadIdx.x;
    int total = NC * Hout * Wout;
    if (idx >= total) return;
    int x = idx % Wout; int t = idx / Wout;
    int y = t % Hout;   int nc = t / Hout;
    const float* ip = in + (size_t)nc * Hin * Win;
    float m = -INFINITY;
    for (int ky = 0; ky < 3; ky++)
        for (int kx = 0; kx < 3; kx++)
            m = fmaxf(m, ip[(2 * y + ky) * Win + (2 * x + kx)]);
    out[idx] = m;
}

// ---------------- ROI pooling: writes padded[n, c*9 + bin] ----------------

__global__ void roi_pool_kernel(const float* __restrict__ feat, const float* __restrict__ bboxes,
                                float* __restrict__ padded)
{
    int idx = blockIdx.x * blockDim.x + threadIdx.x;
    if (idx >= N_BOX * CFEAT) return;
    int c = idx % CFEAT; int n = idx / CFEAT;
    const float* bb = bboxes + (size_t)n * 5;
    const float scale = 13.0f / 224.0f;
    int img = (int)bb[0];
    int sw = (int)rintf(bb[1] * scale);
    int sh = (int)rintf(bb[2] * scale);
    int ew = (int)rintf(bb[3] * scale);
    int eh = (int)rintf(bb[4] * scale);
    float bw = (float)imax(ew - sw + 1, 1) / 3.0f;
    float bh = (float)imax(eh - sh + 1, 1) / 3.0f;
    const float* fp = feat + ((size_t)img * CFEAT + c) * 169;
    float* op = padded + (size_t)n * NOWN + c * 9;
    for (int ph = 0; ph < 3; ph++) {
        int h0 = imin(imax((int)floorf(ph * bh) + sh, 0), 13);
        int h1 = imin(imax((int)ceilf((ph + 1) * bh) + sh, 0), 13);
        for (int pw = 0; pw < 3; pw++) {
            int w0 = imin(imax((int)floorf(pw * bw) + sw, 0), 13);
            int w1 = imin(imax((int)ceilf((pw + 1) * bw) + sw, 0), 13);
            float m = -INFINITY;
            for (int hh = h0; hh < h1; hh++)
                for (int ww = w0; ww < w1; ww++)
                    m = fmaxf(m, fp[hh * 13 + ww]);
            op[ph * 3 + pw] = (h1 > h0 && w1 > w0) ? m : 0.0f;
        }
    }
}

__global__ void zero_pad_row(float* padded)
{
    int f = blockIdx.x * blockDim.x + threadIdx.x;
    if (f < NOWN) padded[(size_t)N_BOX * NOWN + f] = 0.0f;
}

// ---------------- bbox branch (bias dropped: cancels in BN) ----------------

__global__ void bbox_z_kernel(const float* __restrict__ bboxes, const float* __restrict__ bbox_w,
                              float* __restrict__ z)
{
    int idx = blockIdx.x * blockDim.x + threadIdx.x;
    if (idx >= N_BOX * BBH) return;
    int j = idx & 31; int n = idx >> 5;
    const float* bb = bboxes + (size_t)n * 5;
    float w = bb[3] - bb[1], h = bb[4] - bb[2];
    float f5[5] = { bb[1], bb[2], w, h, w / h };
    const float* wr = bbox_w + j * 5;
    float a = 0.0f;
    #pragma unroll
    for (int i = 0; i < 5; i++) a += f5[i] * wr[i];
    z[idx] = a;
}

// ---------------- attention collapse: wk_eff[h] = Wk[h]^T * wk_p[h] ----------------

__global__ void wk_eff_kernel(const float* __restrict__ Wk, const float* __restrict__ attn_w,
                              float* __restrict__ wk_eff)
{
    int idx = blockIdx.x * blockDim.x + threadIdx.x;
    if (idx >= HEADS * NOWN) return;
    int f = idx % NOWN; int hh = idx / NOWN;
    const float* wp = attn_w + hh * (2 * HID) + HID;   // wk_p[h]
    const float* wk = Wk + (size_t)hh * HID * NOWN + f;
    float a = 0.0f;
    for (int d = 0; d < HID; d++) a += wk[(size_t)d * NOWN] * wp[d];
    wk_eff[(size_t)hh * NOWN + f] = a;
}

__global__ void s2_kernel(const float* __restrict__ padded, const float* __restrict__ wk_eff,
                          float* __restrict__ s2)
{
    int idx = blockIdx.x * blockDim.x + threadIdx.x;
    if (idx >= (N_BOX + 1) * HEADS) return;
    int hh = idx & 7; int m = idx >> 3;
    const float* pr = padded + (size_t)m * NOWN;
    const float* we = wk_eff + (size_t)hh * NOWN;
    float a = 0.0f;
    for (int f = 0; f < NOWN; f++) a += pr[f] * we[f];
    s2[idx] = a;
}

// ---------------- softmax over 16 contexts + weighted gather -> ctx (bf16) ----------------

__global__ __launch_bounds__(256) void ctx_kernel(const float* __restrict__ padded,
                                                  const float* __restrict__ s2,
                                                  const int* __restrict__ cin,
                                                  __bf16* __restrict__ ctxBF)
{
    int n = blockIdx.x;
    int tid = threadIdx.x;
    __shared__ int ci[N_CTXI];
    __shared__ float w[HEADS][N_CTXI];
    if (tid < N_CTXI) {
        int c = cin[n * N_CTXI + tid];
        ci[tid] = (c < 0) ? N_BOX : c;
    }
    __syncthreads();
    if (tid < HEADS) {
        float sc[N_CTXI]; float mx = -INFINITY;
        for (int c = 0; c < N_CTXI; c++) { sc[c] = s2[ci[c] * HEADS + tid]; mx = fmaxf(mx, sc[c]); }
        float sum = 0.0f;
        for (int c = 0; c < N_CTXI; c++) { float e = expf(sc[c] - mx); w[tid][c] = e; sum += e; }
        float inv = 1.0f / sum;
        for (int c = 0; c < N_CTXI; c++) w[tid][c] *= inv;
    }
    __syncthreads();
    __bf16* orow = ctxBF + (size_t)n * NCOMB;
    for (int f = tid; f < NOWN; f += 256) {
        float p[N_CTXI];
        #pragma unroll
        for (int c = 0; c < N_CTXI; c++) p[c] = padded[(size_t)ci[c] * NOWN + f];
        for (int hh = 0; hh < HEADS; hh++) {
            float a = 0.0f;
            #pragma unroll
            for (int c = 0; c < N_CTXI; c++) a += w[hh][c] * p[c];
            orow[hh * NOWN + f] = (__bf16)a;
        }
    }
}

// ---------------- helpers: f32 -> bf16 ----------------

__global__ void f32_to_bf16(const float* __restrict__ src, __bf16* __restrict__ dst, size_t n)
{
    size_t i = (size_t)blockIdx.x * blockDim.x + threadIdx.x;
    size_t stride = (size_t)gridDim.x * blockDim.x;
    for (; i < n; i += stride) dst[i] = (__bf16)src[i];
}

__global__ void own_to_combined(const float* __restrict__ padded, __bf16* __restrict__ comb)
{
    size_t i = (size_t)blockIdx.x * blockDim.x + threadIdx.x;
    size_t stride = (size_t)gridDim.x * blockDim.x;
    size_t total = (size_t)N_BOX * NOWN;
    for (; i < total; i += stride) {
        size_t n = i / NOWN, f = i % NOWN;
        comb[n * NDEC + f] = (__bf16)padded[n * NOWN + f];
    }
}

// ---------------- bf16 WMMA GEMM: C(MxN,f32) = A(MxK) * B(NxK)^T ----------------
// Block = 8 waves = 128(M) x 128(N) tile; wave tile 32x64: 2 A-frags x 4 B-frags
// -> 8 accumulators, 8 WMMAs per 6KB of fragment loads (~22 FLOP/B from cache),
// with 4x cross-wave B reuse and 2x A reuse inside the block through L1.
// A fragment: lane holds 16 contiguous K (lanes>=16 offset by 16), row = lane&15.
// B fragment: lane holds 16 contiguous K (lanes>=16 offset by 16), column = lane&15.

__global__ __launch_bounds__(256) void gemm_bf16_wmma(const __bf16* __restrict__ A,
                                                      const __bf16* __restrict__ B,
                                                      float* __restrict__ C,
                                                      int M, int N, int K)
{
    int lane  = threadIdx.x & 31;
    int wave  = threadIdx.x >> 5;
    int waveN = wave & 1;          // 0..1 -> 64-wide N groups
    int waveM = wave >> 1;         // 0..3 -> 32-wide M groups
    int mBase = blockIdx.y * 128 + waveM * 32;
    int nBase = blockIdx.x * 128 + waveN * 64;
    int lr = lane & 15;
    int hi = lane >> 4;            // 0 or 1

    f32x8 acc[2][4];
    #pragma unroll
    for (int im = 0; im < 2; im++)
        #pragma unroll
        for (int t = 0; t < 4; t++)
            acc[im][t] = (f32x8){0.f, 0.f, 0.f, 0.f, 0.f, 0.f, 0.f, 0.f};

    const __bf16* arow0 = A + (size_t)(mBase + lr) * K + hi * 16;
    const __bf16* arow1 = A + (size_t)(mBase + 16 + lr) * K + hi * 16;
    const __bf16* bp[4];
    #pragma unroll
    for (int t = 0; t < 4; t++) {
        int nt = imin(nBase + t * 16 + lr, N - 1);
        bp[t] = B + (size_t)nt * K + hi * 16;
    }

    for (int kb = 0; kb < K; kb += 32) {
        bf16x8 a0lo = *(const bf16x8*)(arow0 + kb);
        bf16x8 a0hi = *(const bf16x8*)(arow0 + kb + 8);
        bf16x8 a1lo = *(const bf16x8*)(arow1 + kb);
        bf16x8 a1hi = *(const bf16x8*)(arow1 + kb + 8);
        bf16x16 af0, af1;
        #pragma unroll
        for (int i = 0; i < 8; i++) {
            af0[i] = a0lo[i]; af0[i + 8] = a0hi[i];
            af1[i] = a1lo[i]; af1[i + 8] = a1hi[i];
        }
        bf16x16 bfv[4];
        #pragma unroll
        for (int t = 0; t < 4; t++)
            bfv[t] = *(const bf16x16*)(bp[t] + kb);

        #pragma unroll
        for (int t = 0; t < 4; t++)
            acc[0][t] = __builtin_amdgcn_wmma_f32_16x16x32_bf16(false, af0, false, bfv[t],
                                                                (short)0, acc[0][t], false, false);
        #pragma unroll
        for (int t = 0; t < 4; t++)
            acc[1][t] = __builtin_amdgcn_wmma_f32_16x16x32_bf16(false, af1, false, bfv[t],
                                                                (short)0, acc[1][t], false, false);
    }

    #pragma unroll
    for (int im = 0; im < 2; im++) {
        int row0 = mBase + im * 16 + hi * 8;
        #pragma unroll
        for (int v = 0; v < 8; v++) {
            float* crow = C + (size_t)(row0 + v) * N;
            #pragma unroll
            for (int t = 0; t < 4; t++) {
                int cc = nBase + t * 16 + lr;
                if (cc < N) crow[cc] = acc[im][t][v];
            }
        }
    }
}

// ---------------- batch-norm (batch stats over rows) + ReLU, column-parallel ----------------

__global__ __launch_bounds__(256) void bn_relu_col(const float* __restrict__ Y, int rows, int cols,
                                                   const float* __restrict__ g, const float* __restrict__ bta,
                                                   float* outF, int ldF, int offF,
                                                   __bf16* outB, int ldB, int offB)
{
    int j = blockIdx.x;
    int tid = threadIdx.x;
    __shared__ float sh_s[256], sh_q[256];
    float s = 0.0f, q = 0.0f;
    for (int r = tid; r < rows; r += 256) {
        float v = Y[(size_t)r * cols + j];
        s += v; q += v * v;
    }
    sh_s[tid] = s; sh_q[tid] = q;
    __syncthreads();
    for (int st = 128; st > 0; st >>= 1) {
        if (tid < st) { sh_s[tid] += sh_s[tid + st]; sh_q[tid] += sh_q[tid + st]; }
        __syncthreads();
    }
    float mean = sh_s[0] / rows;
    float var  = sh_q[0] / rows - mean * mean;
    float sc  = g[j] * rsqrtf(var + 1e-5f);
    float shf = bta[j] - mean * sc;
    for (int r = tid; r < rows; r += 256) {
        float v = fmaxf(Y[(size_t)r * cols + j] * sc + shf, 0.0f);
        if (outF) outF[(size_t)r * ldF + offF + j] = v;
        if (outB) outB[(size_t)r * ldB + offB + j] = (__bf16)v;
    }
}

// ---------------- final classifier ----------------

__global__ void dec2_kernel(const float* __restrict__ h, const float* __restrict__ w2,
                            const float* __restrict__ b2, float* __restrict__ out)
{
    int idx = blockIdx.x * blockDim.x + threadIdx.x;
    if (idx >= N_BOX * NCLS) return;
    int j = idx % NCLS; int n = idx / NCLS;
    const float* hr = h + (size_t)n * NDEC;
    const float* wr = w2 + (size_t)j * NDEC;
    float a = b2[j];
    for (int k = 0; k < NDEC; k++) a += hr[k] * wr[k];
    out[idx] = a;
}

// ---------------- orchestration ----------------

extern "C" void kernel_launch(void* const* d_in, const int* in_sizes, int n_in,
                              void* d_out, int out_size, void* d_ws, size_t ws_size,
                              hipStream_t stream)
{
    const float* images   = (const float*)d_in[0];
    const float* bboxes   = (const float*)d_in[1];
    const int*   ctx_idx  = (const int*)  d_in[2];
    const float* conv1_w  = (const float*)d_in[3];
    const float* conv1_b  = (const float*)d_in[4];
    const float* conv2_w  = (const float*)d_in[5];
    const float* conv2_b  = (const float*)d_in[6];
    const float* conv3_w  = (const float*)d_in[7];
    const float* conv3_b  = (const float*)d_in[8];
    const float* bbox_w   = (const float*)d_in[9];
    /* d_in[10] bbox_b: cancels in BN */
    const float* bn_bb_g  = (const float*)d_in[11];
    const float* bn_bb_b  = (const float*)d_in[12];
    /* d_in[13] Wq, d_in[14] bq: eliminated (softmax shift invariance) */
    const float* Wk       = (const float*)d_in[15];
    /* d_in[16] bk: cancels */
    const float* attn_w   = (const float*)d_in[17];
    /* d_in[18] attn_b: cancels */
    const float* comb_w   = (const float*)d_in[19];
    /* d_in[20] comb_b: cancels in BN */
    const float* bn_cb_g  = (const float*)d_in[21];
    const float* bn_cb_b  = (const float*)d_in[22];
    const float* dec_w1   = (const float*)d_in[23];
    /* d_in[24] dec_b1: cancels in BN */
    const float* bn_dc_g  = (const float*)d_in[25];
    const float* bn_dc_b  = (const float*)d_in[26];
    const float* dec_w2   = (const float*)d_in[27];
    const float* dec_b2   = (const float*)d_in[28];
    float* outp = (float*)d_out;

    // workspace carving (256B aligned)
    size_t off = 0;
    auto take = [&](size_t bytes) {
        size_t o = off;
        off += (bytes + 255) & ~(size_t)255;
        return o;
    };
    char* base = (char*)d_ws;
    float* conv1out   = (float*)(base + take((size_t)4 * 64 * 55 * 55 * 4));
    float* pool1out   = (float*)(base + take((size_t)4 * 64 * 27 * 27 * 4));
    float* conv2out   = (float*)(base + take((size_t)4 * 192 * 27 * 27 * 4));
    float* pool2out   = (float*)(base + take((size_t)4 * 192 * 13 * 13 * 4));
    float* feat       = (float*)(base + take((size_t)4 * 384 * 13 * 13 * 4));
    float* padded     = (float*)(base + take((size_t)(N_BOX + 1) * NOWN * 4));
    float* zbb        = (float*)(base + take((size_t)N_BOX * BBH * 4));
    float* wk_eff     = (float*)(base + take((size_t)HEADS * NOWN * 4));
    float* s2buf      = (float*)(base + take((size_t)(N_BOX + 1) * HEADS * 4));
    __bf16* ctxBF     = (__bf16*)(base + take((size_t)N_BOX * NCOMB * 2));
    __bf16* WBF       = (__bf16*)(base + take((size_t)NOWN * NCOMB * 2));   // reused for dec_w1
    float* Ybuf       = (float*)(base + take((size_t)N_BOX * NDEC * 4));    // GEMM outputs (reused)
    __bf16* combinedBF= (__bf16*)(base + take((size_t)N_BOX * NDEC * 2));
    float* hF32       = (float*)(base + take((size_t)N_BOX * NDEC * 4));
    (void)ws_size; (void)n_in; (void)in_sizes; (void)out_size;

    // 1) backbone
    {
        int t = 4 * 64 * 55 * 55;
        conv2d_bias<<<(t + 255) / 256, 256, 0, stream>>>(images, conv1_w, conv1_b, conv1out,
            4, 3, 224, 224, 64, 55, 55, 11, 11, 4, 2, 1);
    }
    {
        int t = 4 * 64 * 27 * 27;
        maxpool3x3s2<<<(t + 255) / 256, 256, 0, stream>>>(conv1out, pool1out, 4 * 64, 55, 55, 27, 27);
    }
    {
        int t = 4 * 192 * 27 * 27;
        conv2d_bias<<<(t + 255) / 256, 256, 0, stream>>>(pool1out, conv2_w, conv2_b, conv2out,
            4, 64, 27, 27, 192, 27, 27, 5, 5, 1, 2, 1);
    }
    {
        int t = 4 * 192 * 13 * 13;
        maxpool3x3s2<<<(t + 255) / 256, 256, 0, stream>>>(conv2out, pool2out, 4 * 192, 27, 27, 13, 13);
    }
    {
        int t = 4 * 384 * 13 * 13;
        conv2d_bias<<<(t + 255) / 256, 256, 0, stream>>>(pool2out, conv3_w, conv3_b, feat,
            4, 192, 13, 13, 384, 13, 13, 3, 3, 1, 1, 0);
    }

    // 2) own features into padded (1025 x 3488)
    roi_pool_kernel<<<(N_BOX * CFEAT + 255) / 256, 256, 0, stream>>>(feat, bboxes, padded);
    zero_pad_row<<<(NOWN + 255) / 256, 256, 0, stream>>>(padded);
    bbox_z_kernel<<<(N_BOX * BBH + 255) / 256, 256, 0, stream>>>(bboxes, bbox_w, zbb);
    bn_relu_col<<<BBH, 256, 0, stream>>>(zbb, N_BOX, BBH, bn_bb_g, bn_bb_b,
                                         padded, NOWN, NVIS, (__bf16*)0, 0, 0);

    // 3) collapsed attention scores -> softmax weights -> ctx (bf16)
    wk_eff_kernel<<<(HEADS * NOWN + 255) / 256, 256, 0, stream>>>(Wk, attn_w, wk_eff);
    s2_kernel<<<((N_BOX + 1) * HEADS + 255) / 256, 256, 0, stream>>>(padded, wk_eff, s2buf);
    ctx_kernel<<<N_BOX, 256, 0, stream>>>(padded, s2buf, ctx_idx, ctxBF);

    // 4) ctx_rep = relu(BN(ctx @ comb_w^T))  [bias cancels in BN]
    f32_to_bf16<<<8192, 256, 0, stream>>>(comb_w, WBF, (size_t)NOWN * NCOMB);
    {
        dim3 grid((NOWN + 127) / 128, N_BOX / 128);
        gemm_bf16_wmma<<<grid, 256, 0, stream>>>(ctxBF, WBF, Ybuf, N_BOX, NOWN, NCOMB);
    }
    own_to_combined<<<4096, 256, 0, stream>>>(padded, combinedBF);
    bn_relu_col<<<NOWN, 256, 0, stream>>>(Ybuf, N_BOX, NOWN, bn_cb_g, bn_cb_b,
                                          (float*)0, 0, 0, combinedBF, NDEC, NOWN);

    // 5) h = relu(BN(combined @ dec_w1^T))
    f32_to_bf16<<<8192, 256, 0, stream>>>(dec_w1, WBF, (size_t)NDEC * NDEC);
    {
        dim3 grid((NDEC + 127) / 128, N_BOX / 128);
        gemm_bf16_wmma<<<grid, 256, 0, stream>>>(combinedBF, WBF, Ybuf, N_BOX, NDEC, NDEC);
    }
    bn_relu_col<<<NDEC, 256, 0, stream>>>(Ybuf, N_BOX, NDEC, bn_dc_g, bn_dc_b,
                                          hF32, NDEC, 0, (__bf16*)0, 0, 0);

    // 6) logits
    dec2_kernel<<<(N_BOX * NCLS + 255) / 256, 256, 0, stream>>>(hF32, dec_w2, dec_b2, outp);
}
